// ProteinEncoder_558345749244
// MI455X (gfx1250) — matrix operand order repaired and checked by reading, not compiled, not run on hardware
//
#include <hip/hip_runtime.h>

typedef __attribute__((ext_vector_type(16))) _Float16 v16h;
typedef __attribute__((ext_vector_type(8)))  float    v8f;

#define KMAX 15

// ---------------------------------------------------------------------------
// C[M,128] = act( A[M,K] @ W[K,128] + bias )    (M = gridDim.x*64, K % 32 == 0)
// 256 threads = 8 waves. Block owns a 64x128 C tile:
//   wave w -> N cols [16w, 16w+16), 4 M sub-tiles of 16 -> 4 wmma per K step.
// A (64x32) staged row-major f16; W staged TRANSPOSED (Wt[col][k], 128x32 f16)
// so both A and B fragments are contiguous ds_load_b128 pairs.
// ---------------------------------------------------------------------------
__global__ void pe_gemm_bias_act(const float* __restrict__ Am, int lda,
                                 const float* __restrict__ W,
                                 const float* __restrict__ bias,
                                 float* __restrict__ C, int K, int relu) {
  __shared__ _Float16 As[64 * 32];    // [m][k]
  __shared__ _Float16 Wt[128 * 32];   // [col][k]  (transposed W tile)
  const int tid  = threadIdx.x;
  const int lane = tid & 31;
  const int wvi  = tid >> 5;           // 0..7 -> N tile
  const int m0   = blockIdx.x * 64;
  const int hi   = lane >> 4;
  const int col  = wvi * 16 + (lane & 15);
  const int mloc = lane & 15;

  v8f acc[4] = {{}, {}, {}, {}};

  for (int kk = 0; kk < K; kk += 32) {
    // stage A tile: 64x32 f32 = 512 float4 loads (2 per thread)
#pragma unroll
    for (int t = 0; t < 2; ++t) {
      int i = tid + t * 256;                 // 0..511
      int r = i >> 3, c4 = (i & 7) * 4;
      const float4 a4 = *(const float4*)&Am[(size_t)(m0 + r) * lda + kk + c4];
      int base = r * 32 + c4;
      As[base + 0] = (_Float16)a4.x;
      As[base + 1] = (_Float16)a4.y;
      As[base + 2] = (_Float16)a4.z;
      As[base + 3] = (_Float16)a4.w;
    }
    // stage W tile transposed: 32x128 f32 = 1024 float4 loads (4 per thread)
#pragma unroll
    for (int t = 0; t < 4; ++t) {
      int i = tid + t * 256;                 // 0..1023
      int r = i >> 5, c4 = (i & 31) * 4;     // r = k row, c4 = col
      const float4 w4 = *(const float4*)&W[(size_t)(kk + r) * 128 + c4];
      Wt[(c4 + 0) * 32 + r] = (_Float16)w4.x;
      Wt[(c4 + 1) * 32 + r] = (_Float16)w4.y;
      Wt[(c4 + 2) * 32 + r] = (_Float16)w4.z;
      Wt[(c4 + 3) * 32 + r] = (_Float16)w4.w;
    }
    __syncthreads();

    if (kk + 32 < K) {
      __builtin_prefetch(&Am[(size_t)m0 * lda + kk + 32], 0, 1);
      __builtin_prefetch(&W[(size_t)(kk + 32) * 128], 0, 1);
    }

    // B fragment: k = hi*16 + e -> 16 contiguous halves at Wt[col*32 + hi*16]
    v16h bf;
#pragma unroll
    for (int e = 0; e < 16; ++e)
      bf[e] = Wt[col * 32 + (hi << 4) + e];

#pragma unroll
    for (int mt = 0; mt < 4; ++mt) {
      v16h af;
#pragma unroll
      for (int e = 0; e < 16; ++e) {
        int kl = (hi << 4) + e;   // A frag K index
        af[e] = As[(mt * 16 + mloc) * 32 + kl];
      }
      acc[mt] = __builtin_amdgcn_wmma_f32_16x16x32_f16(
          false, af, false, bf, (short)0, acc[mt], false, false);
    }
    __syncthreads();
  }

  const float cb = bias ? bias[col] : 0.0f;
#pragma unroll
  for (int mt = 0; mt < 4; ++mt) {
#pragma unroll
    for (int i = 0; i < 8; ++i) {
      int m = mt * 16 + (hi << 3) + i;
      float v = acc[mt][i] + cb;
      if (relu) v = fmaxf(v, 0.0f);
      C[(size_t)(m0 + m) * 128 + col] = v;
    }
  }
}

// ---------------------------------------------------------------------------
// KNN: one wave per dst point. Per-lane sorted top-KMAX, then 32-way merge.
// ---------------------------------------------------------------------------
__global__ void pe_knn(const float* __restrict__ spos, int n_src,
                       const float* __restrict__ dpos,
                       int k, int exclude_self, int* __restrict__ out_idx) {
  __shared__ float sd[4][32 * KMAX];
  __shared__ int   si[4][32 * KMAX];
  __shared__ int   shead[4][32];
  const int lane = threadIdx.x & 31;
  const int wv   = threadIdx.x >> 5;
  const int dst  = blockIdx.x * 4 + wv;

  const float px = dpos[dst * 3 + 0];
  const float py = dpos[dst * 3 + 1];
  const float pz = dpos[dst * 3 + 2];

  float bd[KMAX]; int bi[KMAX];
#pragma unroll
  for (int j = 0; j < KMAX; ++j) { bd[j] = 3.0e38f; bi[j] = 0x7fffffff; }

  for (int i = lane; i < n_src; i += 32) {
    if (exclude_self && i == dst) continue;
    float dx = px - spos[i * 3 + 0];
    float dy = py - spos[i * 3 + 1];
    float dz = pz - spos[i * 3 + 2];
    float d = dx * dx + dy * dy + dz * dz;
    if (d < bd[KMAX - 1] || (d == bd[KMAX - 1] && i < bi[KMAX - 1])) {
      bd[KMAX - 1] = d; bi[KMAX - 1] = i;
#pragma unroll
      for (int j = KMAX - 1; j > 0; --j) {
        bool sw = (bd[j] < bd[j - 1]) || (bd[j] == bd[j - 1] && bi[j] < bi[j - 1]);
        if (sw) {
          float td = bd[j]; bd[j] = bd[j - 1]; bd[j - 1] = td;
          int ti = bi[j]; bi[j] = bi[j - 1]; bi[j - 1] = ti;
        }
      }
    }
  }
#pragma unroll
  for (int j = 0; j < KMAX; ++j) {
    sd[wv][lane * KMAX + j] = bd[j];
    si[wv][lane * KMAX + j] = bi[j];
  }
  shead[wv][lane] = 0;
  __syncthreads();

  if (lane == 0) {
    for (int j = 0; j < k; ++j) {       // k-way merge of 32 sorted lists
      float best = 3.4e38f; int bidx = 0x7fffffff; int bl = 0;
      for (int l = 0; l < 32; ++l) {
        int hp = shead[wv][l];
        if (hp < KMAX) {
          float d = sd[wv][l * KMAX + hp];
          int id  = si[wv][l * KMAX + hp];
          if (d < best || (d == best && id < bidx)) { best = d; bidx = id; bl = l; }
        }
      }
      out_idx[(size_t)dst * k + j] = bidx;
      shead[wv][bl]++;
    }
  }
}

// ---------------------------------------------------------------------------
// Edge-weight MLP (257 -> 8 -> 1), thread per edge, weights staged in LDS.
// mode==0: i0 = stored[e], i1 = e/k ; mode==1: i0 = e/k, i1 = stored[e]
// ---------------------------------------------------------------------------
__global__ void pe_edge_mlp(const int* __restrict__ stored, int k, int mode,
                            const float* __restrict__ h_a, const float* __restrict__ pos_a,
                            const float* __restrict__ h_b, const float* __restrict__ pos_b,
                            const float* __restrict__ w1, const float* __restrict__ b1,
                            const float* __restrict__ w2, const float* __restrict__ b2,
                            float* __restrict__ ew_out, int E) {
  __shared__ float sm[2073];   // w1(2056) | b1(8) | w2(8) | b2(1)
  for (int t = threadIdx.x; t < 2073; t += blockDim.x) {
    float v;
    if (t < 2056)      v = w1[t];
    else if (t < 2064) v = b1[t - 2056];
    else if (t < 2072) v = w2[t - 2064];
    else               v = b2[0];
    sm[t] = v;
  }
  __syncthreads();

  const int e = blockIdx.x * blockDim.x + threadIdx.x;
  if (e >= E) return;
  const int center = e / k;
  const int st = stored[e];
  const int i0 = mode ? center : st;
  const int i1 = mode ? st : center;

  float dx = pos_a[i0 * 3 + 0] - pos_b[i1 * 3 + 0];
  float dy = pos_a[i0 * 3 + 1] - pos_b[i1 * 3 + 1];
  float dz = pos_a[i0 * 3 + 2] - pos_b[i1 * 3 + 2];
  float d = sqrtf(dx * dx + dy * dy + dz * dz + 1e-12f);

  float acc[8];
#pragma unroll
  for (int j = 0; j < 8; ++j) acc[j] = sm[2056 + j];
  const float* ha = h_a + (size_t)i0 * 128;
  const float* hb = h_b + (size_t)i1 * 128;
  for (int f = 0; f < 128; ++f) {
    float x = ha[f];
#pragma unroll
    for (int j = 0; j < 8; ++j) acc[j] += x * sm[f * 8 + j];
  }
  for (int f = 0; f < 128; ++f) {
    float x = hb[f];
#pragma unroll
    for (int j = 0; j < 8; ++j) acc[j] += x * sm[(128 + f) * 8 + j];
  }
#pragma unroll
  for (int j = 0; j < 8; ++j) acc[j] += d * sm[256 * 8 + j];

  float out = sm[2072];
#pragma unroll
  for (int j = 0; j < 8; ++j) out += fmaxf(acc[j], 0.0f) * sm[2064 + j];
  ew_out[e] = out;
}

// ---------------------------------------------------------------------------
// Graph attention: one wave per (node, head); lane = head dim; online softmax.
// Destination edges are contiguous (fixed degree), so no segment atomics.
// ---------------------------------------------------------------------------
__global__ void pe_attn(const float* __restrict__ q, const float* __restrict__ kb,
                        const float* __restrict__ vb, const float* __restrict__ ew,
                        const int* __restrict__ aa_nbr, const int* __restrict__ av_nbr,
                        const int* __restrict__ vv_nbr,
                        float* __restrict__ outb, int Na, int Nv) {
  const int lane = threadIdx.x & 31;
  const int w = blockIdx.x * (blockDim.x >> 5) + (threadIdx.x >> 5);
  const int node = w >> 2;
  const int head = w & 3;
  const int Eaa = Na * 10, Eav = Nv * 15;

  const float qd = q[(size_t)node * 128 + head * 32 + lane];
  float m = -3.4e38f, s = 0.0f, o = 0.0f;

  int nseg; const int* nb[2]; const float* ewp[2]; int deg[2]; int off[2];
  if (node < Na) {
    nseg = 1;
    nb[0] = aa_nbr + (size_t)node * 10; ewp[0] = ew + (size_t)node * 10;
    deg[0] = 10; off[0] = 0;
  } else {
    int v = node - Na;
    nseg = 2;
    nb[0] = av_nbr + (size_t)v * 15; ewp[0] = ew + Eaa + (size_t)v * 15;
    deg[0] = 15; off[0] = 0;
    nb[1] = vv_nbr + (size_t)v * 15; ewp[1] = ew + Eaa + Eav + (size_t)v * 15;
    deg[1] = 15; off[1] = Na;
  }

  for (int sg = 0; sg < nseg; ++sg) {
    for (int ei = 0; ei < deg[sg]; ++ei) {
      int src = nb[sg][ei] + off[sg];
      float kd = kb[(size_t)src * 128 + head * 32 + lane];
      float dot = qd * kd;
#pragma unroll
      for (int t = 16; t > 0; t >>= 1) dot += __shfl_xor(dot, t, 32);
      dot = dot * 0.17677669529663687f + ewp[sg][ei];   // 1/sqrt(32)
      float mn = fmaxf(m, dot);
      float sc = __expf(m - mn);
      float p  = __expf(dot - mn);
      float vd = vb[(size_t)src * 128 + head * 32 + lane];
      o = o * sc + p * vd;
      s = s * sc + p;
      m = mn;
    }
  }
  outb[(size_t)node * 128 + head * 32 + lane] = o / (s + 1e-9f);
}

// ---------------------------------------------------------------------------
// h = LayerNorm(h + att) with per-layer gamma/beta; one wave per node.
// ---------------------------------------------------------------------------
__global__ void pe_residual_ln(float* __restrict__ h, const float* __restrict__ att,
                               const float* __restrict__ g, const float* __restrict__ b) {
  const int lane = threadIdx.x & 31;
  const int node = blockIdx.x * (blockDim.x >> 5) + (threadIdx.x >> 5);
  float x[4]; float sum = 0.0f, sq = 0.0f;
#pragma unroll
  for (int t = 0; t < 4; ++t) {
    int j = lane + t * 32;
    float v = h[(size_t)node * 128 + j] + att[(size_t)node * 128 + j];
    x[t] = v; sum += v; sq += v * v;
  }
#pragma unroll
  for (int t = 16; t > 0; t >>= 1) {
    sum += __shfl_xor(sum, t, 32);
    sq  += __shfl_xor(sq, t, 32);
  }
  float mean = sum * (1.0f / 128.0f);
  float var  = sq * (1.0f / 128.0f) - mean * mean;
  float inv  = rsqrtf(var + 1e-5f);
#pragma unroll
  for (int t = 0; t < 4; ++t) {
    int j = lane + t * 32;
    h[(size_t)node * 128 + j] = (x[t] - mean) * inv * g[j] + b[j];
  }
}

// Copy [rows,128] block into feat[:, col_off:col_off+128] (feat ld = 640).
__global__ void pe_copy_to_feat(const float* __restrict__ src, float* __restrict__ feat,
                                int col_off) {
  int idx = blockIdx.x * blockDim.x + threadIdx.x;
  int r = idx >> 7, c = idx & 127;
  feat[(size_t)r * 640 + col_off + c] = src[(size_t)r * 128 + c];
}

// ---------------------------------------------------------------------------
extern "C" void kernel_launch(void* const* d_in, const int* in_sizes, int n_in,
                              void* d_out, int out_size, void* d_ws, size_t ws_size,
                              hipStream_t stream) {
  (void)n_in; (void)out_size; (void)ws_size;
  const float* atom_x    = (const float*)d_in[0];
  const float* atom_pos  = (const float*)d_in[1];
  const float* vox_x     = (const float*)d_in[2];
  const float* vox_pos   = (const float*)d_in[3];
  const float* w_atom_in = (const float*)d_in[4];
  const float* b_atom_in = (const float*)d_in[5];
  const float* w_vox_in  = (const float*)d_in[6];
  const float* b_vox_in  = (const float*)d_in[7];
  const float* aa_w1 = (const float*)d_in[8];  const float* aa_b1 = (const float*)d_in[9];
  const float* aa_w2 = (const float*)d_in[10]; const float* aa_b2 = (const float*)d_in[11];
  const float* av_w1 = (const float*)d_in[12]; const float* av_b1 = (const float*)d_in[13];
  const float* av_w2 = (const float*)d_in[14]; const float* av_b2 = (const float*)d_in[15];
  const float* vv_w1 = (const float*)d_in[16]; const float* vv_b1 = (const float*)d_in[17];
  const float* vv_w2 = (const float*)d_in[18]; const float* vv_b2 = (const float*)d_in[19];
  const float* wq   = (const float*)d_in[20];
  const float* wk   = (const float*)d_in[21];
  const float* wv   = (const float*)d_in[22];
  const float* ln_g = (const float*)d_in[23];
  const float* ln_b = (const float*)d_in[24];
  const float* w_o1 = (const float*)d_in[25]; const float* b_o1 = (const float*)d_in[26];
  const float* w_o2 = (const float*)d_in[27]; const float* b_o2 = (const float*)d_in[28];

  const int Na = in_sizes[1] / 3;          // 8192
  const int Nv = in_sizes[3] / 3;          // 4096
  const int N  = Na + Nv;                  // 12288
  const int F  = in_sizes[0] / Na;         // 64
  const int Eaa = Na * 10, Eav = Nv * 15, Evv = Nv * 15;

  float* ws = (float*)d_ws;
  size_t o = 0;
  float* h    = ws + o; o += (size_t)N * 128;
  float* qb   = ws + o; o += (size_t)N * 128;
  float* kb   = ws + o; o += (size_t)N * 128;
  float* vb   = ws + o; o += (size_t)N * 128;
  float* attb = ws + o; o += (size_t)N * 128;
  float* feat = ws + o; o += (size_t)Nv * 640;
  float* hid  = ws + o; o += (size_t)Nv * 128;
  float* ew   = ws + o; o += (size_t)(Eaa + Eav + Evv);
  int* aa_nbr = (int*)(ws + o); o += (size_t)Eaa;
  int* av_nbr = (int*)(ws + o); o += (size_t)Eav;
  int* vv_nbr = (int*)(ws + o); o += (size_t)Evv;

  float* h_vox = h + (size_t)Na * 128;

  // input projections (WMMA GEMMs)
  pe_gemm_bias_act<<<Na / 64, 256, 0, stream>>>(atom_x, F, w_atom_in, b_atom_in, h, F, 0);
  pe_gemm_bias_act<<<Nv / 64, 256, 0, stream>>>(vox_x, F, w_vox_in, b_vox_in, h_vox, F, 0);
  pe_copy_to_feat<<<(Nv * 128) / 256, 256, 0, stream>>>(h_vox, feat, 0);

  // KNN graph construction
  pe_knn<<<Na / 4, 128, 0, stream>>>(atom_pos, Na, atom_pos, 10, 1, aa_nbr);
  pe_knn<<<Nv / 4, 128, 0, stream>>>(atom_pos, Na, vox_pos, 15, 0, av_nbr);
  pe_knn<<<Nv / 4, 128, 0, stream>>>(vox_pos, Nv, vox_pos, 15, 1, vv_nbr);

  // edge-weight MLPs
  pe_edge_mlp<<<Eaa / 256, 256, 0, stream>>>(aa_nbr, 10, 0, h, atom_pos, h, atom_pos,
                                             aa_w1, aa_b1, aa_w2, aa_b2, ew, Eaa);
  pe_edge_mlp<<<Eav / 256, 256, 0, stream>>>(av_nbr, 15, 1, h_vox, vox_pos, h, atom_pos,
                                             av_w1, av_b1, av_w2, av_b2, ew + Eaa, Eav);
  pe_edge_mlp<<<Evv / 256, 256, 0, stream>>>(vv_nbr, 15, 0, h_vox, vox_pos, h_vox, vox_pos,
                                             vv_w1, vv_b1, vv_w2, vv_b2, ew + Eaa + Eav, Evv);

  // 4 blocks x 2 layers of edge-weighted graph attention
  for (int li = 0; li < 8; ++li) {
    const float* wq_l = wq + (size_t)li * 128 * 128;
    const float* wk_l = wk + (size_t)li * 128 * 128;
    const float* wv_l = wv + (size_t)li * 128 * 128;
    pe_gemm_bias_act<<<N / 64, 256, 0, stream>>>(h, 128, wq_l, nullptr, qb, 128, 0);
    pe_gemm_bias_act<<<N / 64, 256, 0, stream>>>(h, 128, wk_l, nullptr, kb, 128, 0);
    pe_gemm_bias_act<<<N / 64, 256, 0, stream>>>(h, 128, wv_l, nullptr, vb, 128, 0);
    pe_attn<<<(N * 4) / 8, 256, 0, stream>>>(qb, kb, vb, ew, aa_nbr, av_nbr, vv_nbr,
                                             attb, Na, Nv);
    pe_residual_ln<<<N / 8, 256, 0, stream>>>(h, attb, ln_g + (size_t)li * 128,
                                              ln_b + (size_t)li * 128);
    if (li & 1) {   // end of block (li = 1,3,5,7) -> snapshot vox features
      pe_copy_to_feat<<<(Nv * 128) / 256, 256, 0, stream>>>(h_vox, feat,
                                                            ((li >> 1) + 1) * 128);
    }
  }

  // output MLP (WMMA GEMMs)
  pe_gemm_bias_act<<<Nv / 64, 256, 0, stream>>>(feat, 640, w_o1, b_o1, hid, 640, 1);
  pe_gemm_bias_act<<<Nv / 64, 256, 0, stream>>>(hid, 128, w_o2, b_o2, (float*)d_out, 128, 0);
}
